// local_emb_D_6597069767219
// MI455X (gfx1250) — matrix operand order repaired and checked by reading, not compiled, not run on hardware
//
#include <hip/hip_runtime.h>

#define HDIM 128

typedef __attribute__((ext_vector_type(16))) _Float16 v16h;
typedef __attribute__((ext_vector_type(8)))  _Float16 v8h;
typedef __attribute__((ext_vector_type(4)))  _Float16 v4h;
typedef __attribute__((ext_vector_type(8)))  float    v8f;

union V16H { v16h v; v8h h[2]; };

// ---------------------------------------------------------------------------
// Phase 1: L2-normalize each row, emit two f16 tables:
//   ewT[row] = (emb[row]/max(||emb[row]||,eps)) * d      (A side)
//   eT [row] =  emb[row]/max(||emb[row]||,eps)           (B side)
// One wave32 per row; each lane owns 4 columns (float4 load, 8B f16 store).
// ---------------------------------------------------------------------------
__global__ void normalize_rows_kernel(const float* __restrict__ emb,
                                      const float* __restrict__ dvec,
                                      _Float16* __restrict__ ewT,
                                      _Float16* __restrict__ eT,
                                      int n_nodes) {
  const int lane = threadIdx.x & 31;
  const int wave = threadIdx.x >> 5;
  const int row  = blockIdx.x * 8 + wave;   // 8 waves per 256-thread block
  if (row >= n_nodes) return;

  const float4* rp = (const float4*)(emb + (size_t)row * HDIM);
  float4 v = rp[lane];                       // columns lane*4 .. lane*4+3
  float ss = v.x * v.x + v.y * v.y + v.z * v.z + v.w * v.w;
  #pragma unroll
  for (int off = 16; off > 0; off >>= 1)
    ss += __shfl_xor(ss, off, 32);

  const float inv = 1.0f / fmaxf(sqrtf(ss), 1e-12f);
  const float4* dp = (const float4*)dvec;
  float4 dd = dp[lane];

  float ex = v.x * inv, ey = v.y * inv, ez = v.z * inv, ew = v.w * inv;

  v4h e4, ew4;
  e4[0]  = (_Float16)ex;          e4[1]  = (_Float16)ey;
  e4[2]  = (_Float16)ez;          e4[3]  = (_Float16)ew;
  ew4[0] = (_Float16)(ex * dd.x); ew4[1] = (_Float16)(ey * dd.y);
  ew4[2] = (_Float16)(ez * dd.z); ew4[3] = (_Float16)(ew * dd.w);

  ((v4h*)(eT  + (size_t)row * HDIM))[lane] = e4;
  ((v4h*)(ewT + (size_t)row * HDIM))[lane] = ew4;
}

// ---------------------------------------------------------------------------
// Phase 2: per-wave batch of 16 edges.
//   A = ewT[src[e0..e15]] (16 x K), B = eT[dst[e0..e15]]^T (K x 16)
//   C += A*B via v_wmma_f32_16x16x32_f16 over K=128 (4 WMMAs), take diag.
// VGPR layouts per CDNA5 ISA 7.12.2 (16-bit A 16x32, 16-bit B 32x16, f32 C).
// ---------------------------------------------------------------------------
__global__ void edge_dot_wmma_kernel(const _Float16* __restrict__ ewT,
                                     const _Float16* __restrict__ eT,
                                     const int* __restrict__ src,
                                     const int* __restrict__ dst,
                                     const float* __restrict__ scale,
                                     float* __restrict__ out,
                                     int n_edges) {
  const int lane = threadIdx.x & 31;
  const int wave = threadIdx.x >> 5;
  const long long eBase = ((long long)blockIdx.x * 8 + wave) * 16;

  const int m  = lane & 15;   // A row / B column owned by this lane
  const int hi = lane >> 4;   // half-wave selects K sub-range

  long long eIdx = eBase + m;
  if (eIdx >= n_edges) eIdx = n_edges - 1;   // clamp; store is masked later

  const int s = src[eIdx];
  const int t = dst[eIdx];
  const _Float16* aRow = ewT + (size_t)s * HDIM;
  const _Float16* bRow = eT  + (size_t)t * HDIM;

  v8f c = {};
  #pragma unroll
  for (int k0 = 0; k0 < HDIM; k0 += 32) {
    V16H a, b;
    // A 16x32 f16: lanes 0-15 -> K {k0+0..7, k0+16..23}; lanes 16-31 -> +8
    a.h[0] = *(const v8h*)(aRow + k0 + hi * 8);
    a.h[1] = *(const v8h*)(aRow + k0 + 16 + hi * 8);
    // B 32x16 f16: lanes 0-15 -> K {k0+0..7, k0+16..23}; lanes 16-31 -> +8
    b.h[0] = *(const v8h*)(bRow + k0 + hi * 8);
    b.h[1] = *(const v8h*)(bRow + k0 + 16 + hi * 8);
    c = __builtin_amdgcn_wmma_f32_16x16x32_f16(
        /*neg_a=*/false, a.v, /*neg_b=*/false, b.v,
        /*c_mod=*/(short)0, c, /*reuse_a=*/false, /*reuse_b=*/false);
  }

  // Diagonal extraction from f32 C layout:
  //   m = 0..7  lives at VGPR m, lane m        (lanes 0-7)
  //   m = 8..15 lives at VGPR m-8, lane m+16   (lanes 24-31)
  const int r = lane & 7;
  float dv = c[0];
  dv = (r == 1) ? c[1] : dv;
  dv = (r == 2) ? c[2] : dv;
  dv = (r == 3) ? c[3] : dv;
  dv = (r == 4) ? c[4] : dv;
  dv = (r == 5) ? c[5] : dv;
  dv = (r == 6) ? c[6] : dv;
  dv = (r == 7) ? c[7] : dv;

  const bool low  = lane < 8;
  const bool high = lane >= 24;
  const int  mOut = low ? lane : (lane - 16);
  const long long oIdx = eBase + mOut;
  if ((low || high) && oIdx < n_edges)
    out[oIdx] = dv * scale[0];
}

// ---------------------------------------------------------------------------
extern "C" void kernel_launch(void* const* d_in, const int* in_sizes, int n_in,
                              void* d_out, int out_size, void* d_ws, size_t ws_size,
                              hipStream_t stream) {
  const float* emb   = (const float*)d_in[0];
  const int*   src   = (const int*)d_in[1];
  const int*   dst   = (const int*)d_in[2];
  const float* dvec  = (const float*)d_in[3];
  const float* scale = (const float*)d_in[4];
  float*       out   = (float*)d_out;

  const int n_nodes = in_sizes[0] / HDIM;
  const int n_edges = in_sizes[1];

  _Float16* ewT = (_Float16*)d_ws;                       // 25.6 MB
  _Float16* eT  = ewT + (size_t)n_nodes * HDIM;          // +25.6 MB

  // Phase 1: 8 rows (waves) per 256-thread block
  {
    int blocks = (n_nodes + 7) / 8;
    normalize_rows_kernel<<<blocks, 256, 0, stream>>>(emb, dvec, ewT, eT, n_nodes);
  }
  // Phase 2: 16 edges per wave, 8 waves per block -> 128 edges per block
  {
    long long waves  = ((long long)n_edges + 15) / 16;
    int       blocks = (int)((waves + 7) / 8);
    edge_dot_wmma_kernel<<<blocks, 256, 0, stream>>>(ewT, eT, src, dst, scale,
                                                     out, n_edges);
  }
}
